// SparseAutoencoder_58325655880075
// MI455X (gfx1250) — matrix operand order repaired and checked by reading, not compiled, not run on hardware
//
#include <hip/hip_runtime.h>
#include <hip/hip_bf16.h>

// ---------------------------------------------------------------------------
// TopK Sparse Autoencoder forward for MI455X (gfx1250, wave32, WMMA + TDM).
//
// Encoder GEMM (8192x16384x2048, ~550 GFLOP) on v_wmma_f32_16x16x32_f16 with
// f32 accumulation; 2x4 register tile per wave (8 WMMA / 6 v16h loads per
// K-step) with explicit register double-buffering so global loads overlap the
// matrix pipe. Decoder is sparse (k=64) gather-accumulate. pre-activations
// are staged in d_out's z region, then the per-row radix-select top-k kernel
// DMAs each 64KB row into LDS with TENSOR_LOAD_TO_LDS and rewrites it.
// ---------------------------------------------------------------------------

typedef __attribute__((ext_vector_type(16))) _Float16 v16h;
typedef __attribute__((ext_vector_type(8)))  float    v8f;
typedef __attribute__((ext_vector_type(4)))  unsigned u32x4;
typedef __attribute__((ext_vector_type(8)))  int      i32x8;
typedef __attribute__((ext_vector_type(4)))  int      i32x4;

#define D_IN   2048
#define D_HID  16384
#define N_TOK  8192
#define TOPK   64

// tile counts
#define KT   64     // 2048 / 32   K tiles
#define MT   512    // 8192 / 16   M tiles (A)
#define NT   1024   // 16384 / 16  N tiles (B)

#if __has_builtin(__builtin_amdgcn_tensor_load_to_lds) && __has_builtin(__builtin_amdgcn_s_wait_tensorcnt)
#define HAVE_TDM 1
#else
#define HAVE_TDM 0
#endif

// ---------------------------------------------------------------------------
// Pack A = (x - b_dec) into f16 WMMA A-operand tiles (ISA 7.12.2 layout).
// ---------------------------------------------------------------------------
__global__ __launch_bounds__(256) void pack_A_kernel(const float* __restrict__ x,
                                                     const float* __restrict__ b_dec,
                                                     v16h* __restrict__ Ap) {
    int gid  = blockIdx.x * 256 + threadIdx.x;
    int lane = gid & 31;
    int tile = gid >> 5;                          // tm*KT + tk
    int tk   = tile & (KT - 1);
    int tm   = tile >> 6;
    int M     = tm * 16 + (lane & 15);
    int kbase = tk * 32 + ((lane & 16) ? 8 : 0);
    const float* xrow = x + (size_t)M * D_IN;
    v16h v;
#pragma unroll
    for (int j = 0; j < 16; ++j) {
        int K = kbase + (j < 8 ? j : j + 8);
        v[j] = (_Float16)(xrow[K] - b_dec[K]);
    }
    Ap[(size_t)tile * 32 + lane] = v;
}

// ---------------------------------------------------------------------------
// Pack B = W_enc^T into f16 WMMA B-operand tiles: lane holds N=tn*16+(lane&15),
// K(j) = 8*(lane>=16) + (j < 8 ? j : j + 8)  => two 8-wide f32 runs from a W_enc row.
// ---------------------------------------------------------------------------
__global__ __launch_bounds__(256) void pack_B_kernel(const float* __restrict__ Wenc,
                                                     v16h* __restrict__ Bp) {
    int gid  = blockIdx.x * 256 + threadIdx.x;
    int lane = gid & 31;
    int tile = gid >> 5;                          // tn*KT + tk
    int tk   = tile & (KT - 1);
    int tn   = tile >> 6;
    int h  = tn * 16 + (lane & 15);
    int kb = tk * 32 + ((lane & 16) ? 8 : 0);
    const float* wrow = Wenc + (size_t)h * D_IN + kb;
    v16h v;
#pragma unroll
    for (int j = 0; j < 16; ++j) v[j] = (_Float16)wrow[j < 8 ? j : j + 8];
    Bp[(size_t)tile * 32 + lane] = v;
}

// ---------------------------------------------------------------------------
// Transpose W_dec [2048 x 16384] -> W_decT [16384 x 2048] (f32), LDS-tiled.
// ---------------------------------------------------------------------------
__global__ __launch_bounds__(256) void transpose_wdec_kernel(const float* __restrict__ W,
                                                             float* __restrict__ WT) {
    __shared__ float tile[32][33];
    int bx = blockIdx.x;              // over D_HID/32
    int by = blockIdx.y;              // over D_IN/32
    int tx = threadIdx.x;             // 0..31
    int ty = threadIdx.y;             // 0..7
#pragma unroll
    for (int i = 0; i < 4; ++i) {
        int r = by * 32 + ty + i * 8;
        tile[ty + i * 8][tx] = W[(size_t)r * D_HID + bx * 32 + tx];
    }
    __syncthreads();
#pragma unroll
    for (int i = 0; i < 4; ++i) {
        int h = bx * 32 + ty + i * 8;
        WT[(size_t)h * D_IN + by * 32 + tx] = tile[tx][ty + i * 8];
    }
}

// ---------------------------------------------------------------------------
// Encoder GEMM: block = 8 waves arranged 4(M) x 2(N); block tile 128x128.
// Wave tile: 32(M) x 64(N) = 8 accumulators; K loop double-buffered in regs.
// ---------------------------------------------------------------------------
#define WMMA_F16(A, B, C) __builtin_amdgcn_wmma_f32_16x16x32_f16(false, A, false, B, (short)0, C, false, false)

__global__ __launch_bounds__(256) void enc_gemm_kernel(const v16h* __restrict__ Ap,
                                                       const v16h* __restrict__ Bp,
                                                       const float* __restrict__ b_enc,
                                                       float* __restrict__ pre) {
    int lane = threadIdx.x & 31;
    int wave = threadIdx.x >> 5;   // 0..7
    int wm   = wave & 3;           // M group (0..3)
    int wn   = wave >> 2;          // N group (0..1)
    int bn   = blockIdx.x;         // 0..127
    int bm   = blockIdx.y;         // 0..63

    int tm0 = bm * 8 + wm * 2;     // two consecutive A tiles
    int tnb = bn * 8 + wn * 4;     // four consecutive B tiles

    const v16h* ap0 = Ap + (size_t)tm0 * (KT * 32) + lane;
    const v16h* ap1 = ap0 + (size_t)(KT * 32);
    const v16h* bp0 = Bp + (size_t)(tnb + 0) * (KT * 32) + lane;
    const v16h* bp1 = Bp + (size_t)(tnb + 1) * (KT * 32) + lane;
    const v16h* bp2 = Bp + (size_t)(tnb + 2) * (KT * 32) + lane;
    const v16h* bp3 = Bp + (size_t)(tnb + 3) * (KT * 32) + lane;

    v8f a00 = {}, a01 = {}, a02 = {}, a03 = {};
    v8f a10 = {}, a11 = {}, a12 = {}, a13 = {};

    // prologue: load K-step 0
    v16h ca0 = ap0[0], ca1 = ap1[0];
    v16h cb0 = bp0[0], cb1 = bp1[0], cb2 = bp2[0], cb3 = bp3[0];

    for (int tk = 0; tk < KT - 1; ++tk) {
        int nk = (tk + 1) * 32;
        // issue next-iteration loads into fresh registers (overlap with WMMA)
        v16h na0 = ap0[nk];
        v16h na1 = ap1[nk];
        v16h nb0 = bp0[nk];
        v16h nb1 = bp1[nk];
        v16h nb2 = bp2[nk];
        v16h nb3 = bp3[nk];

        a00 = WMMA_F16(ca0, cb0, a00);
        a01 = WMMA_F16(ca0, cb1, a01);
        a02 = WMMA_F16(ca0, cb2, a02);
        a03 = WMMA_F16(ca0, cb3, a03);
        a10 = WMMA_F16(ca1, cb0, a10);
        a11 = WMMA_F16(ca1, cb1, a11);
        a12 = WMMA_F16(ca1, cb2, a12);
        a13 = WMMA_F16(ca1, cb3, a13);

        ca0 = na0; ca1 = na1;
        cb0 = nb0; cb1 = nb1; cb2 = nb2; cb3 = nb3;
    }
    // epilogue K-step
    a00 = WMMA_F16(ca0, cb0, a00);
    a01 = WMMA_F16(ca0, cb1, a01);
    a02 = WMMA_F16(ca0, cb2, a02);
    a03 = WMMA_F16(ca0, cb3, a03);
    a10 = WMMA_F16(ca1, cb0, a10);
    a11 = WMMA_F16(ca1, cb1, a11);
    a12 = WMMA_F16(ca1, cb2, a12);
    a13 = WMMA_F16(ca1, cb3, a13);

    // C/D layout: acc[r] at (M = r + 8*(lane>=16), N = lane&15)
    int col  = bn * 128 + wn * 64 + (lane & 15);
    int row0 = bm * 128 + wm * 32 + ((lane & 16) ? 8 : 0);
    float be0 = b_enc[col +  0];
    float be1 = b_enc[col + 16];
    float be2 = b_enc[col + 32];
    float be3 = b_enc[col + 48];
#pragma unroll
    for (int r = 0; r < 8; ++r) {
        float* o0 = pre + (size_t)(row0 + r) * D_HID + col;
        o0[ 0] = a00[r] + be0;
        o0[16] = a01[r] + be1;
        o0[32] = a02[r] + be2;
        o0[48] = a03[r] + be3;
        float* o1 = pre + (size_t)(row0 + 16 + r) * D_HID + col;
        o1[ 0] = a10[r] + be0;
        o1[16] = a11[r] + be1;
        o1[32] = a12[r] + be2;
        o1[48] = a13[r] + be3;
    }
}

// ---------------------------------------------------------------------------
// Per-row top-k (k=64): TDM the 64KB row into LDS, 4-pass radix select on
// monotone float keys, ties broken by lowest index, rewrite row as z in place,
// emit compact (idx,val) list + per-row l1/l0 partials.
// Dynamic-LDS layout (row at offset 0 to match D#.lds_addr = 0):
//   [0      .. 65536) row     16384 f32
//   [65536  .. 66560) hist    256 u32
//   [66560  .. 67584) red     256 i32/f32 (reused)
//   [67584  .. 68096) eqb     128 i32
//   [68096  .. 68352) chosen  64 i32
//   [68352  .. 68384) scalars (prefix, need, neq, nchosen, total)
// ---------------------------------------------------------------------------
#define TOPK_SMEM 68384

__device__ __forceinline__ unsigned fkey(float f) {
    unsigned u = __float_as_uint(f);
    return (u & 0x80000000u) ? ~u : (u | 0x80000000u);
}

__global__ __launch_bounds__(256) void topk_kernel(float* __restrict__ z,
                                                   int*   __restrict__ cnt,
                                                   int*   __restrict__ idxb,
                                                   float* __restrict__ valb,
                                                   float* __restrict__ l1row) {
    extern __shared__ char smem[];
    float*    row    = (float*)smem;                          // 16384 f32
    unsigned* hist   = (unsigned*)(smem + 65536);             // 256
    int*      red    = (int*)(smem + 66560);                  // 256 (reused)
    int*      eqb    = (int*)(smem + 67584);                  // 128
    int*      chosen = (int*)(smem + 68096);                  // 64
    int*      scal   = (int*)(smem + 68352);                  // 5 ints

    int t  = threadIdx.x;
    int r0 = blockIdx.x;
    float* zr = z + (size_t)r0 * D_HID;

#if HAVE_TDM
    // Stage the row into LDS with the Tensor Data Mover: 1-row tile of
    // 16384 x f32. Issued once by wave 0; TENSORcnt waited by the issuing
    // wave, then a workgroup barrier publishes LDS to all waves.
    if (t < 32) {
        unsigned long long ga = (unsigned long long)(uintptr_t)zr;
        u32x4 g0;
        g0[0] = 1u;                                           // count=1, user mode
        g0[1] = 0u;                                           // lds_addr = 0
        g0[2] = (unsigned)(ga & 0xFFFFFFFFull);               // global_addr lo
        g0[3] = (unsigned)((ga >> 32) & 0x1FFFFFFull) | (2u << 30); // addr hi | type=2
        i32x8 g1;
        g1[0] = (int)(2u << 16);      // workgroup_mask=0, data_size=4B
        g1[1] = (int)0x40000000;      // tensor_dim0 = 16384 (bits[63:48])
        g1[2] = (int)(1u << 16);      // tensor_dim0 hi = 0, tensor_dim1 = 1
        g1[3] = (int)0x40000000;      // tile_dim0 = 16384 (bits[127:112])
        g1[4] = 0;                    // tile_dim1 = 0 (unused), tile_dim2 = 0
        g1[5] = (int)D_HID;           // tensor_dim0_stride lo
        g1[6] = 0;
        g1[7] = 0;
        i32x4 g2 = {0, 0, 0, 0};
        i32x4 g3 = {0, 0, 0, 0};
#if defined(__clang_major__) && (__clang_major__ >= 23)
        i32x8 g4 = {0, 0, 0, 0, 0, 0, 0, 0};
        __builtin_amdgcn_tensor_load_to_lds(g0, g1, g2, g3, g4, 0);
#else
        __builtin_amdgcn_tensor_load_to_lds(g0, g1, g2, g3, 0);
#endif
        __builtin_amdgcn_s_wait_tensorcnt(0);
    }
    __syncthreads();
#else
    {
        float4* row4 = (float4*)row;
        const float4* zr4 = (const float4*)zr;
        for (int i = t; i < D_HID / 4; i += 256) row4[i] = zr4[i];
        __syncthreads();
    }
#endif
    if (t == 0) { scal[0] = 0; scal[1] = TOPK; scal[2] = 0; }
    __syncthreads();

    // 4-pass radix select (MSB first) for the key of the k-th largest
    for (int pass = 0; pass < 4; ++pass) {
        int shift = 24 - 8 * pass;
        hist[t] = 0;
        __syncthreads();
        unsigned pfx = (unsigned)scal[0];
        for (int i = t; i < D_HID; i += 256) {
            unsigned key = fkey(row[i]);
            if (pass == 0 || (key >> (shift + 8)) == pfx)
                atomicAdd(&hist[(key >> shift) & 255u], 1u);
        }
        __syncthreads();
        if (t == 0) {
            unsigned need = (unsigned)scal[1], cum = 0;
            int b = 255;
            for (; b > 0; --b) {
                if (cum + hist[b] >= need) break;
                cum += hist[b];
            }
            scal[1] = (int)(need - cum);
            scal[0] = (int)(((unsigned)scal[0] << 8) | (unsigned)b);
        }
        __syncthreads();
    }
    unsigned T = (unsigned)scal[0];   // exact key of threshold value
    int need   = scal[1];             // how many key==T entries to take

    // gather key==T indices (cap 128)
    for (int i = t; i < D_HID; i += 256) {
        if (fkey(row[i]) == T) {
            int p = atomicAdd(&scal[2], 1);
            if (p < 128) eqb[p] = i;
        }
    }
    __syncthreads();
    if (t == 0) {
        int ne = scal[2] < 128 ? scal[2] : 128;
        int nc = need < ne ? need : ne;
        for (int a = 0; a < nc; ++a) {            // smallest indices first
            int best = a;
            for (int b2 = a + 1; b2 < ne; ++b2)
                if (eqb[b2] < eqb[best]) best = b2;
            int tmp = eqb[a]; eqb[a] = eqb[best]; eqb[best] = tmp;
            chosen[a] = eqb[a];
        }
        scal[3] = nc;
    }
    __syncthreads();
    int nchosen = scal[3];

    // write z row (relu-masked), keep masked values in LDS, l1 partial
    float l1loc = 0.f;
    for (int i = t; i < D_HID; i += 256) {
        float f = row[i];
        unsigned key = fkey(f);
        bool take = key > T;
        if (!take && key == T) {
            for (int j = 0; j < nchosen; ++j)
                if (chosen[j] == i) { take = true; break; }
        }
        float zv = (take && f > 0.f) ? f : 0.f;
        zr[i]  = zv;
        row[i] = zv;
        l1loc += zv;
    }
    __syncthreads();

    // deterministic compaction: thread t owns indices [64t, 64t+64)
    int base = t * 64;
    int c = 0;
    for (int j = 0; j < 64; ++j) c += (row[base + j] > 0.f) ? 1 : 0;
    red[t] = c;
    __syncthreads();
    if (t == 0) {
        int run = 0;
        for (int i = 0; i < 256; ++i) { int cc = red[i]; red[i] = run; run += cc; }
        scal[4] = run;                 // <= 64
    }
    __syncthreads();
    int off = red[t];
    int rowbase = r0 * TOPK;
    for (int j = 0; j < 64; ++j) {
        float v = row[base + j];
        if (v > 0.f) { idxb[rowbase + off] = base + j; valb[rowbase + off] = v; ++off; }
    }
    __syncthreads();
    ((float*)red)[t] = l1loc;
    __syncthreads();
    if (t == 0) {
        float s = 0.f;
        for (int i = 0; i < 256; ++i) s += ((float*)red)[i];
        l1row[r0] = s;
        cnt[r0]   = scal[4];
    }
}

// ---------------------------------------------------------------------------
// Sparse decoder: x_hat[row] = sum_j val_j * W_decT[idx_j] + b_dec.
// ---------------------------------------------------------------------------
__global__ __launch_bounds__(256) void dec_kernel(const float* __restrict__ x,
                                                  const float* __restrict__ WdecT,
                                                  const float* __restrict__ b_dec,
                                                  const int*   __restrict__ cnt,
                                                  const int*   __restrict__ idxb,
                                                  const float* __restrict__ valb,
                                                  float* __restrict__ xhat,
                                                  float* __restrict__ reconrow) {
    __shared__ int   s_i[TOPK];
    __shared__ float s_v[TOPK];
    __shared__ float redf[256];
    int t  = threadIdx.x;
    int r0 = blockIdx.x;
    int n  = cnt[r0];
    if (t < n) { s_i[t] = idxb[r0 * TOPK + t]; s_v[t] = valb[r0 * TOPK + t]; }
    __syncthreads();

    float acc[8];
#pragma unroll
    for (int i = 0; i < 8; ++i) acc[i] = b_dec[t + 256 * i];
    for (int j = 0; j < n; ++j) {
        const float* wr = WdecT + (size_t)s_i[j] * D_IN;
        float v = s_v[j];
#pragma unroll
        for (int i = 0; i < 8; ++i) acc[i] += v * wr[t + 256 * i];
    }
    const float* xr = x + (size_t)r0 * D_IN;
    float* orow = xhat + (size_t)r0 * D_IN;
    float ss = 0.f;
#pragma unroll
    for (int i = 0; i < 8; ++i) {
        int d = t + 256 * i;
        orow[d] = acc[i];
        float df = xr[d] - acc[i];
        ss += df * df;
    }
    redf[t] = ss;
    __syncthreads();
    if (t == 0) {
        float s = 0.f;
        for (int i = 0; i < 256; ++i) s += redf[i];
        reconrow[r0] = s;
    }
}

// ---------------------------------------------------------------------------
// Deterministic final reduction of the 3 scalar metrics.
// ---------------------------------------------------------------------------
__global__ __launch_bounds__(256) void finalize_kernel(const float* __restrict__ l1row,
                                                       const float* __restrict__ reconrow,
                                                       const int*   __restrict__ cnt,
                                                       float* __restrict__ scal) {
    __shared__ float r1[256], r2[256];
    __shared__ int   r3[256];
    int t = threadIdx.x;
    float a = 0.f, b = 0.f; int c = 0;
    for (int i = t * 32; i < t * 32 + 32; ++i) { a += l1row[i]; b += reconrow[i]; c += cnt[i]; }
    r1[t] = a; r2[t] = b; r3[t] = c;
    __syncthreads();
    if (t == 0) {
        float s1 = 0.f, s2 = 0.f; int s3 = 0;
        for (int i = 0; i < 256; ++i) { s1 += r1[i]; s2 += r2[i]; s3 += r3[i]; }
        scal[0] = s2 / (float)((size_t)N_TOK * D_IN);     // recon_loss
        scal[1] = s1 / ((float)N_TOK * (float)D_HID);     // l1_loss
        scal[2] = (float)s3 / (float)N_TOK;               // l0
    }
}

// ---------------------------------------------------------------------------
extern "C" void kernel_launch(void* const* d_in, const int* in_sizes, int n_in,
                              void* d_out, int out_size, void* d_ws, size_t ws_size,
                              hipStream_t stream) {
    (void)in_sizes; (void)n_in; (void)out_size; (void)ws_size;
    const float* x     = (const float*)d_in[0];
    const float* W_enc = (const float*)d_in[1];
    const float* b_enc = (const float*)d_in[2];
    const float* W_dec = (const float*)d_in[3];
    const float* b_dec = (const float*)d_in[4];
    // d_in[5] = k (always 64 here)

    // workspace layout (bytes)
    char* ws = (char*)d_ws;
    v16h*  Ap       = (v16h*)(ws + 0);                       //  32 MB
    v16h*  Bp       = (v16h*)(ws + 33554432ull);             //  64 MB
    float* WdecT    = (float*)(ws + 100663296ull);           // 128 MB
    int*   idxb     = (int*)  (ws + 234881024ull);           //   2 MB
    float* valb     = (float*)(ws + 236978176ull);           //   2 MB
    int*   cnt      = (int*)  (ws + 239075328ull);           //  32 KB
    float* l1row    = (float*)(ws + 239108096ull);           //  32 KB
    float* reconrow = (float*)(ws + 239140864ull);           //  32 KB

    // d_out layout: x_hat [8192x2048] | z [8192x16384] | recon, l1, l0
    float* xhat = (float*)d_out;
    float* z    = xhat + (size_t)N_TOK * D_IN;
    float* scal = z + (size_t)N_TOK * D_HID;

    // 1) pack operands (L2-resident f16 panels) + transpose decoder weight
    pack_A_kernel<<<MT * KT * 32 / 256, 256, 0, stream>>>(x, b_dec, Ap);
    pack_B_kernel<<<NT * KT * 32 / 256, 256, 0, stream>>>(W_enc, Bp);
    transpose_wdec_kernel<<<dim3(D_HID / 32, D_IN / 32), dim3(32, 8), 0, stream>>>(W_dec, WdecT);

    // 2) encoder GEMM -> pre staged in d_out's z region
    enc_gemm_kernel<<<dim3(D_HID / 128, N_TOK / 128), 256, 0, stream>>>(Ap, Bp, b_enc, z);

    // 3) per-row top-k: overwrites pre with z in place, emits compact list
    topk_kernel<<<N_TOK, 256, TOPK_SMEM, stream>>>(z, cnt, idxb, valb, l1row);

    // 4) sparse decoder + recon partials
    dec_kernel<<<N_TOK, 256, 0, stream>>>(x, WdecT, b_dec, cnt, idxb, valb, xhat, reconrow);

    // 5) scalar metrics
    finalize_kernel<<<1, 256, 0, stream>>>(l1row, reconrow, cnt, scal);
}
